// MoELinear_15994458210634
// MI455X (gfx1250) — matrix operand order repaired and checked by reading, not compiled, not run on hardware
//
#include <hip/hip_runtime.h>

#define BROWS 8192
#define DIN   1024
#define DOUT  1024
#define NE    16
#define KTOT  (NE * 1024)  // 16384

#define BM 128
#define BN 128
#define BK 64
#define NKT (KTOT / BK)    // 256 k-tiles
#define ASTR 72            // padded LDS row stride (halves) -> 144B
#define BSTR 72

typedef _Float16 v16h __attribute__((ext_vector_type(16)));
typedef float    v8f  __attribute__((ext_vector_type(8)));
typedef unsigned int u32x4 __attribute__((ext_vector_type(4)));
typedef int          i32x8 __attribute__((ext_vector_type(8)));
typedef int          i32x4 __attribute__((ext_vector_type(4)));

static __device__ __forceinline__ unsigned short f2h(float f) {
    _Float16 h = (_Float16)f;
    unsigned short u;
    __builtin_memcpy(&u, &h, 2);
    return u;
}

// ---------------------------------------------------------------------------
// Kernel 1: gate logits + softmax.  One block (256 thr) per row of x.
// ---------------------------------------------------------------------------
__global__ __launch_bounds__(256) void moe_gate_kernel(
    const float* __restrict__ x, const float* __restrict__ Wg,
    const float* __restrict__ bg, float* __restrict__ g) {
    __shared__ float red[256 * 16];
    const int row = blockIdx.x;
    const int tid = threadIdx.x;

    float acc[16];
#pragma unroll
    for (int e = 0; e < 16; ++e) acc[e] = 0.f;

    const float* xr = x + (size_t)row * DIN;
    for (int i = tid; i < DIN; i += 256) {
        float xv = xr[i];
        const float* wr = Wg + (size_t)i * 16;
#pragma unroll
        for (int e = 0; e < 16; ++e) acc[e] += xv * wr[e];
    }
#pragma unroll
    for (int e = 0; e < 16; ++e) red[tid * 16 + e] = acc[e];
    __syncthreads();
    for (int s = 128; s > 0; s >>= 1) {
        if (tid < s) {
#pragma unroll
            for (int e = 0; e < 16; ++e)
                red[tid * 16 + e] += red[(tid + s) * 16 + e];
        }
        __syncthreads();
    }
    if (tid == 0) {
        float v[16];
        float m = -3.0e38f;
#pragma unroll
        for (int e = 0; e < 16; ++e) { v[e] = red[e] + bg[e]; m = fmaxf(m, v[e]); }
        float s = 0.f;
#pragma unroll
        for (int e = 0; e < 16; ++e) { v[e] = __expf(v[e] - m); s += v[e]; }
        float inv = 1.f / s;
#pragma unroll
        for (int e = 0; e < 16; ++e) g[(size_t)row * 16 + e] = v[e] * inv;
    }
}

// ---------------------------------------------------------------------------
// Kernel 2: We fp32 [KTOT][DOUT] -> Wbt f16 [DOUT][KTOT] (transposed, N-major)
// ---------------------------------------------------------------------------
__global__ __launch_bounds__(256) void moe_wconv_kernel(
    const float* __restrict__ We, _Float16* __restrict__ Wbt) {
    __shared__ _Float16 t[32][33];
    const int k0 = blockIdx.x * 32;
    const int o0 = blockIdx.y * 32;
    const int tid = threadIdx.x;
    const int lr = tid >> 5;
    const int lc = tid & 31;
#pragma unroll
    for (int p = 0; p < 4; ++p) {
        int kr = p * 8 + lr;
        t[kr][lc] = (_Float16)We[(size_t)(k0 + kr) * DOUT + o0 + lc];
    }
    __syncthreads();
#pragma unroll
    for (int p = 0; p < 4; ++p) {
        int orow = p * 8 + lr;
        Wbt[(size_t)(o0 + orow) * KTOT + k0 + lc] = t[lc][orow];
    }
}

// ---------------------------------------------------------------------------
// Kernel 3: fused MoE GEMM with TDM double-buffered B tiles.
// out[b,o] = sum_k A'(b,k)*Wbt[o,k] + sum_e g[b,e]*be[e,o]
// A'(b,k) = g[b, k>>10] * x[b, k&1023]
// ---------------------------------------------------------------------------
__global__ __launch_bounds__(256) void moe_gemm_kernel(
    const float* __restrict__ x, const float* __restrict__ g,
    const _Float16* __restrict__ Wbt, const float* __restrict__ be,
    float* __restrict__ out) {
    __shared__ __align__(16) unsigned short As[BM * ASTR];        // 18432 B
    __shared__ __align__(16) unsigned short Bs[2][BN * BSTR];     // 36864 B
    __shared__ float gS[BM * 16];                                 //  8192 B

    const int tid  = threadIdx.x;
    const int o0   = blockIdx.x * BN;
    const int b0   = blockIdx.y * BM;
    const int wave = tid >> 5;
    const int lane = tid & 31;
    const int wm   = wave & 1;   // M direction: 2 waves x 64 rows
    const int wn   = wave >> 1;  // N direction: 4 waves x 32 cols
    const int lm   = lane & 15;
    const int hi   = lane >> 4;

    const int arow = tid >> 4;        // A staging row base (0..15)
    const int acol = (tid & 15) * 4;  // A staging col (float4)

    // ---- TDM issue helper: B tile (128 rows x 64 halves) -> Bs[buf] ----
    // D# built per CDNA5 ISA ch.8: 2B elements, row stride KTOT, LDS padding
    // of 4 DWORDs every 32 DWORDs reproduces the 72-half padded row stride.
    auto tdm_issue_b = [&](int kq, int buf) {
        unsigned long long ga =
            (unsigned long long)(const void*)(Wbt + (size_t)o0 * KTOT + kq);
        unsigned int lds = (unsigned int)(unsigned long long)(const void*)(&Bs[buf][0]);
        u32x4 g0v = { 1u,                                  // count=1, user D#
                      lds,                                 // lds_addr
                      (unsigned int)ga,                    // global_addr[31:0]
                      (unsigned int)((ga >> 32) & 0x01FFFFFFu) | (2u << 30) }; // addr[56:32] | type=2
        i32x8 g1v = { (int)((1u << 16) |                   // data_size = 2B
                            (1u << 20) |                   // pad_enable
                            (4u << 22) |                   // pad_interval: 32 DWORDs
                            (3u << 25)),                   // pad_amount: 4 DWORDs
                      (int)(0x4000u << 16),                // tensor_dim0 = 16384 (lo16)
                      (int)(0x400u << 16),                 // tensor_dim1 = 1024 (lo16)
                      (int)(64u << 16),                    // tile_dim0 = 64
                      (int)128,                            // tile_dim1 = 128, tile_dim2 = 0
                      (int)16384,                          // tensor_dim0_stride lo32
                      0, 0 };
        i32x4 zz4 = { 0, 0, 0, 0 };
        i32x8 zz8 = { 0, 0, 0, 0, 0, 0, 0, 0 };
        __builtin_amdgcn_tensor_load_to_lds(g0v, g1v, zz4, zz4, zz8, 0);
    };

    // ---- prologue ----
    for (int idx = tid; idx < BM * 16; idx += 256)
        gS[idx] = g[(size_t)b0 * 16 + idx];

    if (wave == 0) tdm_issue_b(0, 0);

    float4 areg[8];
#pragma unroll
    for (int p = 0; p < 8; ++p) {
        int row = p * 16 + arow;
        areg[p] = *reinterpret_cast<const float4*>(x + (size_t)(b0 + row) * DIN + acol);
    }
    __syncthreads();   // gS visible before A store uses it
#pragma unroll
    for (int p = 0; p < 8; ++p) {
        int row = p * 16 + arow;
        float gv = gS[row * 16 + 0];
        unsigned lo  = (unsigned)f2h(areg[p].x * gv) | ((unsigned)f2h(areg[p].y * gv) << 16);
        unsigned hiw = (unsigned)f2h(areg[p].z * gv) | ((unsigned)f2h(areg[p].w * gv) << 16);
        *reinterpret_cast<uint2*>(&As[row * ASTR + acol]) = make_uint2(lo, hiw);
    }
    if (wave == 0) __builtin_amdgcn_s_wait_tensorcnt(0);
    __syncthreads();

    v8f acc[4][2];
#pragma unroll
    for (int i = 0; i < 4; ++i)
#pragma unroll
        for (int j = 0; j < 2; ++j)
#pragma unroll
            for (int r = 0; r < 8; ++r) acc[i][j][r] = 0.f;

    union Frag { v16h v; uint4 q[2]; };

    // ---- main pipelined loop ----
    for (int kt = 0; kt < NKT; ++kt) {
        const int knext = (kt + 1) * BK;
        const bool more = (kt + 1) < NKT;

        if (more) {
            if (wave == 0) tdm_issue_b(knext, (kt + 1) & 1);
            const int i0n = knext & (DIN - 1);
#pragma unroll
            for (int p = 0; p < 8; ++p) {
                int row = p * 16 + arow;
                areg[p] = *reinterpret_cast<const float4*>(
                    x + (size_t)(b0 + row) * DIN + i0n + acol);
            }
        }

        // ---- compute tile kt from LDS ----
        const unsigned short* bsp = &Bs[kt & 1][0];
#pragma unroll
        for (int ks = 0; ks < 2; ++ks) {
            const int kk = ks * 32;
            Frag a[4], b2[2];
#pragma unroll
            for (int mt = 0; mt < 4; ++mt) {
                int row = wm * 64 + mt * 16 + lm;
                int c0  = kk + hi * 8;
                a[mt].q[0] = *reinterpret_cast<const uint4*>(&As[row * ASTR + c0]);
                a[mt].q[1] = *reinterpret_cast<const uint4*>(&As[row * ASTR + c0 + 16]);
            }
#pragma unroll
            for (int nt = 0; nt < 2; ++nt) {
                int orow = wn * 32 + nt * 16 + lm;
                int c0   = kk + hi * 8;
                b2[nt].q[0] = *reinterpret_cast<const uint4*>(&bsp[orow * BSTR + c0]);
                b2[nt].q[1] = *reinterpret_cast<const uint4*>(&bsp[orow * BSTR + c0 + 16]);
            }
#pragma unroll
            for (int mt = 0; mt < 4; ++mt)
#pragma unroll
                for (int nt = 0; nt < 2; ++nt)
                    acc[mt][nt] = __builtin_amdgcn_wmma_f32_16x16x32_f16(
                        false, a[mt].v, false, b2[nt].v,
                        (short)0, acc[mt][nt], false, false);
        }

        __syncthreads();   // all waves done reading As / Bs[kt&1]

        if (more) {
            const int en = knext >> 10;   // expert for next K-block
#pragma unroll
            for (int p = 0; p < 8; ++p) {
                int row = p * 16 + arow;
                float gv = gS[row * 16 + en];
                unsigned lo  = (unsigned)f2h(areg[p].x * gv) | ((unsigned)f2h(areg[p].y * gv) << 16);
                unsigned hiw = (unsigned)f2h(areg[p].z * gv) | ((unsigned)f2h(areg[p].w * gv) << 16);
                *reinterpret_cast<uint2*>(&As[row * ASTR + acol]) = make_uint2(lo, hiw);
            }
            if (wave == 0) __builtin_amdgcn_s_wait_tensorcnt(0);
            __syncthreads();
        }
    }

    // ---- epilogue: gate-weighted bias + store ----
#pragma unroll
    for (int nt = 0; nt < 2; ++nt) {
        const int col = wn * 32 + nt * 16 + lm;
        float bev[16];
#pragma unroll
        for (int e = 0; e < 16; ++e)
            bev[e] = be[(size_t)e * DOUT + o0 + col];
#pragma unroll
        for (int mt = 0; mt < 4; ++mt) {
#pragma unroll
            for (int r = 0; r < 8; ++r) {
                int row = wm * 64 + mt * 16 + hi * 8 + r;
                float bias = 0.f;
#pragma unroll
                for (int e = 0; e < 16; ++e)
                    bias += gS[row * 16 + e] * bev[e];
                out[(size_t)(b0 + row) * DOUT + o0 + col] = acc[mt][nt][r] + bias;
            }
        }
    }
}

// ---------------------------------------------------------------------------
extern "C" void kernel_launch(void* const* d_in, const int* in_sizes, int n_in,
                              void* d_out, int out_size, void* d_ws, size_t ws_size,
                              hipStream_t stream) {
    const float* x  = (const float*)d_in[0];
    const float* Wg = (const float*)d_in[1];
    const float* bg = (const float*)d_in[2];
    const float* We = (const float*)d_in[3];
    const float* be = (const float*)d_in[4];
    float* out = (float*)d_out;

    char* ws = (char*)d_ws;
    float*    g   = (float*)ws;                      // 512 KB
    _Float16* Wbt = (_Float16*)(ws + (1 << 19));     // 32 MB

    moe_gate_kernel<<<BROWS, 256, 0, stream>>>(x, Wg, bg, g);

    dim3 gw(KTOT / 32, DOUT / 32);
    moe_wconv_kernel<<<gw, 256, 0, stream>>>(We, Wbt);

    dim3 gg(DOUT / BN, BROWS / BM);
    moe_gemm_kernel<<<gg, 256, 0, stream>>>(x, g, Wbt, be, out);
}
